// MultiHeadAttention_12764642803857
// MI455X (gfx1250) — matrix operand order repaired and checked by reading, not compiled, not run on hardware
//
#include <hip/hip_runtime.h>

typedef __attribute__((ext_vector_type(16))) __bf16    v16bf;
typedef __attribute__((ext_vector_type(8)))  float     v8f;
typedef __attribute__((ext_vector_type(4)))  unsigned  v4u;
typedef __attribute__((ext_vector_type(8)))  int       v8i;
typedef __attribute__((ext_vector_type(4)))  int       v4i;

#define B_      32
#define T_      4096
#define D_      1024
#define H_      16
#define DEPTH_  64
#define TM      32            // t-rows per block
#define KC      32            // k-chunk == WMMA K
#define NCHUNK  (D_ / KC)     // 32
#define NTILES  (T_ / TM)     // 128 t-tiles

__device__ __forceinline__ unsigned bfbits(float x) {   // f32 -> bf16 bits, RNE
    unsigned u = __builtin_bit_cast(unsigned, x);
    return (u + 0x7FFFu + ((u >> 16) & 1u)) >> 16;
}
__device__ __forceinline__ __bf16 f2bf(float x) {
    unsigned short s = (unsigned short)bfbits(x);
    return __builtin_bit_cast(__bf16, s);
}

// ---------------------------------------------------------------- kernel 0
// One-time: Wv [k][n] f32  ->  wvT [n][k] bf16 (2 MB, L2-resident afterwards)
__global__ void wv_convert_kernel(const float* __restrict__ Wv,
                                  __bf16* __restrict__ wvT) {
    __shared__ float tile[64][65];
    int k0 = blockIdx.x * 64, n0 = blockIdx.y * 64, tid = threadIdx.x;
    for (int i = tid; i < 64 * 64; i += 256) {
        int r = i >> 6, c = i & 63;
        tile[r][c] = Wv[(size_t)(k0 + r) * D_ + n0 + c];
    }
    __syncthreads();
    for (int i = tid; i < 64 * 64; i += 256) {
        int r = i >> 6, c = i & 63;
        wvT[(size_t)(n0 + r) * D_ + k0 + c] = f2bf(tile[c][r]);
    }
}

// ---------------------------------------------------------------- kernel 1
// q = state @ Wq + bq          [32,1024]
__global__ void qproj_kernel(const float* __restrict__ state,
                             const float* __restrict__ Wq,
                             const float* __restrict__ bq,
                             float* __restrict__ qbuf) {
    int b = blockIdx.x, tid = threadIdx.x;
    const float* st = state + (size_t)b * D_;
    for (int j = tid; j < D_; j += 256) {
        float s = bq[j];
        for (int i = 0; i < D_; ++i) s = fmaf(st[i], Wq[(size_t)i * D_ + j], s);
        qbuf[(size_t)b * D_ + j] = s;
    }
}

// ---------------------------------------------------------------- kernel 2
// Fused: TDM-staged A, L2-direct bf16 B, WMMA v-tile, logits, softmax partials.
union SMem {
    float af32[2][TM * KC];      // double-buffered features tile (TDM dest) 8 KB
    float vtile[TM * D_];        // v tile (fp32)                           128 KB
};

__device__ __forceinline__ void tdm_load_tile(const float* gsrc, unsigned lds_off) {
    unsigned long long ga = (unsigned long long)(const void*)gsrc;
    // D# group 0: count=1 | lds_addr | 57-bit global addr | type=2
    v4u g0 = { 1u, lds_off, (unsigned)ga,
               (unsigned)((ga >> 32) & 0x01FFFFFFu) | (2u << 30) };
    // D# group 1: data_size=4B; tensor_dim0/1=32; tile_dim0/1=32; dim0_stride=1024
    v8i g1 = { (int)0x00020000,        // workgroup_mask=0, data_size=2 (4B)
               (int)(32u << 16),       // tensor_dim0[15:0] in [31:16]
               (int)(32u << 16),       // tensor_dim1[15:0] in [31:16]
               (int)(32u << 16),       // tile_dim0 in [31:16]
               32,                     // tile_dim1 in [15:0], tile_dim2=0
               (int)1024,              // tensor_dim0_stride[31:0]
               0, 0 };
    v4i z4 = {0, 0, 0, 0};
#if __has_include(<hip/amd_detail/amd_gfx1250_TDM.h>)
    v8i z8 = {0, 0, 0, 0, 0, 0, 0, 0};
    __builtin_amdgcn_tensor_load_to_lds(g0, g1, z4, z4, z8, 0);
#else
    __builtin_amdgcn_tensor_load_to_lds(g0, g1, z4, z4, 0);
#endif
}

__global__ __launch_bounds__(256)
void fused_attn_kernel(const float* __restrict__ features,
                       const __bf16* __restrict__ wvT,
                       const float* __restrict__ bv,
                       const float* __restrict__ qbuf,
                       float* __restrict__ mbuf,
                       float* __restrict__ sbuf,
                       float* __restrict__ obuf) {
    __shared__ SMem sm;
    __shared__ float slog[H_ * TM];
    __shared__ float smax[H_], ssum[H_];

    const int tile  = blockIdx.x;
    const int b     = blockIdx.y;
    const int tid   = threadIdx.x;
    const int wave  = tid >> 5;
    const int lane  = tid & 31;
    const int lane16 = lane & 15;
    const int g     = lane >> 4;
    const int trow0 = tile * TM;
    const int nbase = wave * 128;          // each wave owns 128 N columns
    const float* atile0 = features + ((size_t)b * T_ + trow0) * D_;

    const v8f vzero = {0.f, 0.f, 0.f, 0.f, 0.f, 0.f, 0.f, 0.f};
    v8f acc[2][8];
    for (int mt = 0; mt < 2; ++mt)
        for (int nt = 0; nt < 8; ++nt) acc[mt][nt] = vzero;

    // prime the TDM pipeline: chunk 0 -> buffer 0   (wave 0 drives the DMA)
    if (tid < 32)
        tdm_load_tile(atile0, (unsigned)(uintptr_t)&sm.af32[0][0]);

    for (int kc = 0; kc < NCHUNK; ++kc) {
        const int k0 = kc * KC;
        const int cur = kc & 1;
        if (tid < 32) {
            if (kc + 1 < NCHUNK)       // issue next chunk into the other buffer
                tdm_load_tile(atile0 + k0 + KC,
                              (unsigned)(uintptr_t)&sm.af32[cur ^ 1][0]);
            if (kc + 1 < NCHUNK) __builtin_amdgcn_s_wait_tensorcnt(1);
            else                 __builtin_amdgcn_s_wait_tensorcnt(0);
        }
        __syncthreads();               // current A tile visible to all waves

        // A fragments: read f32 pairs from LDS, convert+pack to bf16
        union AF { v16bf v; unsigned u[8]; } af[2];
        for (int mt = 0; mt < 2; ++mt) {
            const float2* ap =
                (const float2*)&sm.af32[cur][(mt * 16 + lane16) * KC];
            #pragma unroll
            for (int p = 0; p < 8; ++p) {
                int kp = g * 16 + p * 2;
                float2 t = ap[kp >> 1];
                af[mt].u[p] = bfbits(t.x) | (bfbits(t.y) << 16);
            }
        }
        // B fragments straight from L2-resident pre-transposed bf16 Wv
        #pragma unroll
        for (int nt = 0; nt < 8; ++nt) {
            union BF { v16bf v; uint4 q[2]; } bfr;
            const uint4* bp = (const uint4*)
                (wvT + (size_t)(nbase + nt * 16 + lane16) * D_ + k0 + g * 16);
            bfr.q[0] = bp[0];
            bfr.q[1] = bp[1];
            #pragma unroll
            for (int mt = 0; mt < 2; ++mt) {
                acc[mt][nt] = __builtin_amdgcn_wmma_f32_16x16x32_bf16(
                    false, af[mt].v, false, bfr.v, (short)0, acc[mt][nt],
                    false, false);
            }
        }
        if (k0 + KC < D_)   // cheap L2 prefetch hint for next B slab
            __builtin_prefetch(wvT + (size_t)(nbase + lane16) * D_ + k0 + KC, 0, 1);
        __syncthreads();               // release A buffer before TDM overwrites it
    }

    // ---- spill v tile (+bv) to LDS (overwrites TDM buffers)
    float bvv[8];
    #pragma unroll
    for (int nt = 0; nt < 8; ++nt) bvv[nt] = bv[nbase + nt * 16 + lane16];
    for (int mt = 0; mt < 2; ++mt)
        for (int nt = 0; nt < 8; ++nt) {
            int col = nbase + nt * 16 + lane16;
            #pragma unroll
            for (int r = 0; r < 8; ++r) {
                int row = mt * 16 + g * 8 + r;   // C/D layout: VGPR r, lane-half g
                sm.vtile[row * D_ + col] = acc[mt][nt][r] + bvv[nt];
            }
        }
    __syncthreads();

    // ---- logits: l[h][t] = (q_h . v[t,h*64:]) / 8
    const float* qb = qbuf + (size_t)b * D_;
    for (int idx = tid; idx < H_ * TM; idx += 256) {
        int h = idx >> 5, t = idx & 31;
        const float* vp = &sm.vtile[t * D_ + h * DEPTH_];
        const float* qp = qb + h * DEPTH_;
        float s = 0.f;
        for (int d = 0; d < DEPTH_; ++d) s = fmaf(qp[d], vp[d], s);
        slog[idx] = s * 0.125f;
    }
    __syncthreads();

    // ---- per-head partial max / expsum
    if (tid < H_) {
        float m = -3.0e38f;
        for (int t = 0; t < TM; ++t) m = fmaxf(m, slog[tid * TM + t]);
        float s = 0.f;
        for (int t = 0; t < TM; ++t) {
            float e = __expf(slog[tid * TM + t] - m);
            slog[tid * TM + t] = e;
            s += e;
        }
        smax[tid] = m; ssum[tid] = s;
        mbuf[((size_t)b * NTILES + tile) * H_ + tid] = m;
        sbuf[((size_t)b * NTILES + tile) * H_ + tid] = s;
    }
    __syncthreads();

    // ---- partial weighted v: o[h][d] = sum_t e[h][t] * v[t][h*64+d]
    for (int idx = tid; idx < H_ * DEPTH_; idx += 256) {
        int h = idx >> 6, d = idx & 63;
        float o = 0.f;
        for (int t = 0; t < TM; ++t)
            o = fmaf(slog[h * TM + t], sm.vtile[t * D_ + h * DEPTH_ + d], o);
        obuf[(((size_t)b * NTILES + tile) * H_ + h) * DEPTH_ + d] = o;
    }
    (void)smax; (void)ssum;
}

// ---------------------------------------------------------------- kernel 3
__global__ void reduce_kernel(const float* __restrict__ mbuf,
                              const float* __restrict__ sbuf,
                              const float* __restrict__ obuf,
                              float* __restrict__ pooled) {
    __shared__ float red[NTILES];
    __shared__ float wfac[NTILES];
    int h = blockIdx.x, b = blockIdx.y, t = threadIdx.x;

    float m = mbuf[((size_t)b * NTILES + t) * H_ + h];
    red[t] = m;
    __syncthreads();
    for (int s = NTILES / 2; s > 0; s >>= 1) {
        if (t < s) red[t] = fmaxf(red[t], red[t + s]);
        __syncthreads();
    }
    float M = red[0];
    __syncthreads();

    float w = __expf(m - M);
    wfac[t] = w;
    red[t] = sbuf[((size_t)b * NTILES + t) * H_ + h] * w;
    __syncthreads();
    for (int s = NTILES / 2; s > 0; s >>= 1) {
        if (t < s) red[t] += red[t + s];
        __syncthreads();
    }
    float S = red[0];
    __syncthreads();

    if (t < DEPTH_) {
        float o = 0.f;
        for (int tt = 0; tt < NTILES; ++tt)
            o = fmaf(wfac[tt],
                     obuf[(((size_t)b * NTILES + tt) * H_ + h) * DEPTH_ + t], o);
        pooled[(size_t)b * D_ + h * DEPTH_ + t] = o / S;
    }
}

// ---------------------------------------------------------------- kernel 4
__global__ void dense_kernel(const float* __restrict__ pooled,
                             const float* __restrict__ Wd,
                             const float* __restrict__ bd,
                             float* __restrict__ out) {
    int b = blockIdx.x, tid = threadIdx.x;
    const float* p = pooled + (size_t)b * D_;
    for (int j = tid; j < D_; j += 256) {
        float s = bd[j];
        for (int i = 0; i < D_; ++i) s = fmaf(p[i], Wd[(size_t)i * D_ + j], s);
        out[(size_t)b * D_ + j] = s;
    }
}

// ---------------------------------------------------------------- launch
extern "C" void kernel_launch(void* const* d_in, const int* in_sizes, int n_in,
                              void* d_out, int out_size, void* d_ws, size_t ws_size,
                              hipStream_t stream) {
    const float* features = (const float*)d_in[0];
    const float* state    = (const float*)d_in[1];
    const float* Wq       = (const float*)d_in[2];
    const float* bq       = (const float*)d_in[3];
    const float* Wv       = (const float*)d_in[4];
    const float* bv       = (const float*)d_in[5];
    const float* Wd       = (const float*)d_in[6];
    const float* bd       = (const float*)d_in[7];

    char*   wsb    = (char*)d_ws;
    __bf16* wvT    = (__bf16*)wsb;                                 // 2 MB
    float*  qbuf   = (float*)(wsb + (size_t)2 * 1024 * 1024);
    float*  mbuf   = qbuf + (size_t)B_ * D_;
    float*  sbuf   = mbuf + (size_t)B_ * NTILES * H_;
    float*  obuf   = sbuf + (size_t)B_ * NTILES * H_;
    float*  pooled = obuf + (size_t)B_ * NTILES * H_ * DEPTH_;

    wv_convert_kernel<<<dim3(16, 16), 256, 0, stream>>>(Wv, wvT);
    qproj_kernel<<<B_, 256, 0, stream>>>(state, Wq, bq, qbuf);
    fused_attn_kernel<<<dim3(NTILES, B_), 256, 0, stream>>>(
        features, wvT, bv, qbuf, mbuf, sbuf, obuf);
    reduce_kernel<<<dim3(H_, B_), 128, 0, stream>>>(mbuf, sbuf, obuf, pooled);
    dense_kernel<<<B_, 256, 0, stream>>>(pooled, Wd, bd, (float*)d_out);
}
